// IALSHSelfAttention_76570676953298
// MI455X (gfx1250) — matrix operand order repaired and checked by reading, not compiled, not run on hardware
//
#include <hip/hip_runtime.h>
#include <hip/hip_bf16.h>

// ---------------------------------------------------------------------------
// IALSH self-attention, fused flash-style, CDNA5 (gfx1250) wave32 WMMA bf16.
// B=2, L=2048, D=768, H=12, DH=64, M_LSH=2, U=0.75
// ---------------------------------------------------------------------------

#define BB   2
#define LL   2048
#define DD   768
#define HH   12
#define DHD  64
#define BHN  (BB * HH)           // 24
#define NROW (BHN * LL)          // 49152

typedef __bf16 v16bf __attribute__((ext_vector_type(16)));
typedef __bf16 v8bf  __attribute__((ext_vector_type(8)));
typedef float  v8f   __attribute__((ext_vector_type(8)));

// --- WMMA wrapper: D(f32 16x16) = A(bf16 16x32) * B(bf16 32x16) + C ---------
__device__ __forceinline__ v8f wmma_bf16(v16bf a, v16bf b, v8f c) {
  return __builtin_amdgcn_wmma_f32_16x16x32_bf16(
      /*neg_a=*/false, a, /*neg_b=*/false, b,
      /*c_mod=*/(short)0, c, /*reuse_a=*/false, /*reuse_b=*/false);
}

// --- A fragment (16x32 bf16): lane holds row (caller bakes row into base),
//     elements K = kbase + hi*16 + {0..7} and kbase + hi*16 + 8 + {0..7} ----
__device__ __forceinline__ v16bf load_a_frag_bf16(const __bf16* row_base,
                                                  int kbase, int hi) {
  const __bf16* p = row_base + kbase + hi * 16;
  v8bf lo = *(const v8bf*)(p);
  v8bf hh = *(const v8bf*)(p + 8);
  v16bf r;
#pragma unroll
  for (int i = 0; i < 8; ++i) { r[i] = lo[i]; r[i + 8] = hh[i]; }
  return r;
}

// --- B fragment (32x16 bf16): lane column = lane&15, k rows contiguous:
//     element B[k,n] read from mem[(row0 + lane&15)*ld + kbase + hi*16 + k'] -
__device__ __forceinline__ v16bf load_b_frag_bf16(const __bf16* base, int ld,
                                                  int row0, int kbase, int lane) {
  const int c = lane & 15, hi = lane >> 4;
  return *(const v16bf*)(base + (size_t)(row0 + c) * ld + kbase + hi * 16);
}

__global__ void init_kernel(unsigned int* mmax_bits) {
  if (threadIdx.x == 0 && blockIdx.x == 0) *mmax_bits = 0u;
}

// ---------------------------------------------------------------------------
// K0: bulk f32 -> bf16 conversion (X and both weight matrices), 8 elems/thread
// so the main GEMM loops never touch f32 operands.
// ---------------------------------------------------------------------------
__global__ void cvt_bf16_kernel(const float* __restrict__ src,
                                __bf16* __restrict__ dst, int n) {
  const int i = (blockIdx.x * blockDim.x + threadIdx.x) * 8;
  if (i >= n) return;
  v8f x = *(const v8f*)(src + i);
  v8bf y;
#pragma unroll
  for (int j = 0; j < 8; ++j) y[j] = (__bf16)x[j];
  *(v8bf*)(dst + i) = y;
}

// ---------------------------------------------------------------------------
// K1: qk = X @ Wqk^T + bqk ; v = X @ Wv^T + bv  (pure bf16 WMMA, f32 accum)
// One wave -> 16 token rows x one head (64 cols). Software-pipelined k-loop:
// fragments for step k+1 are issued before the WMMAs of step k, so the 8
// matrix ops cover the 9 fragment loads of the next step (double buffering via
// rotated registers; unroll-by-2 turns the rotation into renaming).
// ---------------------------------------------------------------------------
__global__ void __launch_bounds__(256) proj_kernel(
    const __bf16* __restrict__ Xb,    // [B*L, D] bf16
    const __bf16* __restrict__ Wqkb,  // [D, D]   bf16
    const float*  __restrict__ bqk,   // [D]
    const __bf16* __restrict__ Wvb,   // [D, D]   bf16
    const float*  __restrict__ bv,    // [D]
    __bf16* __restrict__ qk_bf,       // [BH, L, 64]
    __bf16* __restrict__ vT_bf,       // [BH, 64, L]   (transposed!)
    float* __restrict__ norm2,        // [BH, L]
    unsigned int* __restrict__ mmax_bits) {
  const int lane = threadIdx.x & 31;
  const int wave = blockIdx.x * (blockDim.x >> 5) + (threadIdx.x >> 5);
  const int mt = wave / HH;          // token-row tile: 0..255
  const int h  = wave % HH;          // head
  const int hi = lane >> 4;
  const int cl = lane & 15;

  v8f accq[4] = {};
  v8f accv[4] = {};

  const __bf16* xrow = Xb + (size_t)(mt * 16 + cl) * DD;

  // ---- prologue: fill current buffers for kb = 0 ----
  v16bf a_cur = load_a_frag_bf16(xrow, 0, hi);
  v16bf bq_cur[4], bw_cur[4];
#pragma unroll
  for (int j = 0; j < 4; ++j)
    bq_cur[j] = load_b_frag_bf16(Wqkb, DD, h * 64 + j * 16, 0, lane);
#pragma unroll
  for (int j = 0; j < 4; ++j)
    bw_cur[j] = load_b_frag_bf16(Wvb, DD, h * 64 + j * 16, 0, lane);

#pragma unroll 2
  for (int kb = 0; kb < DD; kb += 32) {
    // Issue next step's loads first (kn wraps harmlessly on the last step).
    const int kn = (kb + 32 < DD) ? (kb + 32) : 0;
    v16bf a_nxt = load_a_frag_bf16(xrow, kn, hi);
    v16bf bq_nxt[4], bw_nxt[4];
#pragma unroll
    for (int j = 0; j < 4; ++j)
      bq_nxt[j] = load_b_frag_bf16(Wqkb, DD, h * 64 + j * 16, kn, lane);
#pragma unroll
    for (int j = 0; j < 4; ++j)
      bw_nxt[j] = load_b_frag_bf16(Wvb, DD, h * 64 + j * 16, kn, lane);
    // Matrix ops on the already-resident current buffers.
#pragma unroll
    for (int j = 0; j < 4; ++j) accq[j] = wmma_bf16(a_cur, bq_cur[j], accq[j]);
#pragma unroll
    for (int j = 0; j < 4; ++j) accv[j] = wmma_bf16(a_cur, bw_cur[j], accv[j]);
    // Rotate buffers.
    a_cur = a_nxt;
#pragma unroll
    for (int j = 0; j < 4; ++j) { bq_cur[j] = bq_nxt[j]; bw_cur[j] = bw_nxt[j]; }
  }

  float biasq[4], biasv[4];
#pragma unroll
  for (int j = 0; j < 4; ++j) {
    biasq[j] = bqk[h * 64 + j * 16 + cl];
    biasv[j] = bv[h * 64 + j * 16 + cl];
  }

  float wavemax = 0.0f;
#pragma unroll
  for (int r = 0; r < 8; ++r) {
    const int row = mt * 16 + r + 8 * hi;   // global token row in [0, B*L)
    const int b = row >> 11;                // /2048
    const int l = row & 2047;
    const int bh = b * HH + h;
    float s2 = 0.0f;
#pragma unroll
    for (int j = 0; j < 4; ++j) {
      float q = accq[j][r] + biasq[j];
      float v = accv[j][r] + biasv[j];
      s2 += q * q;
      qk_bf[((size_t)bh * LL + l) * 64 + j * 16 + cl] = (__bf16)q;
      vT_bf[((size_t)bh * 64 + j * 16 + cl) * LL + l] = (__bf16)v;
    }
#pragma unroll
    for (int mm = 1; mm < 16; mm <<= 1) s2 += __shfl_xor(s2, mm, 32);
    if (cl == 0) norm2[(size_t)bh * LL + l] = s2;
    wavemax = fmaxf(wavemax, s2);
  }
#pragma unroll
  for (int mm = 1; mm < 32; mm <<= 1)
    wavemax = fmaxf(wavemax, __shfl_xor(wavemax, mm, 32));
  if (lane == 0) atomicMax(mmax_bits, __float_as_uint(wavemax));
}

// ---------------------------------------------------------------------------
// K2: build P = [qk*U/Mmax, 0.5-n^2, 0.5-n^4, pad0] (K padded 66->96),
//     a_bf16 padded to 96, and Qdot = (qk*U/Mmax) . a[:,:64].
// ---------------------------------------------------------------------------
__global__ void prep_kernel(const float* __restrict__ a_in,     // [NROW, 66]
                            const __bf16* __restrict__ qk_bf,   // [NROW, 64]
                            const float* __restrict__ norm2,    // [NROW]
                            const unsigned int* __restrict__ mmax_bits,
                            __bf16* __restrict__ qkP,           // [NROW, 96]
                            __bf16* __restrict__ a_bf,          // [NROW, 96]
                            float* __restrict__ qdot) {         // [NROW]
  const int idx = blockIdx.x * blockDim.x + threadIdx.x;
  if (idx >= NROW) return;
  const float mmax = sqrtf(__uint_as_float(*mmax_bits));
  const float scale = 0.75f / mmax;

  const __bf16* qrow = qk_bf + (size_t)idx * 64;
  __bf16* prow = qkP + (size_t)idx * 96;
  __bf16* arow = a_bf + (size_t)idx * 96;
  const float* ain = a_in + (size_t)idx * 66;

  float qd = 0.0f;
#pragma unroll 8
  for (int d = 0; d < 64; ++d) {
    float qs = (float)qrow[d] * scale;
    prow[d] = (__bf16)qs;
    qd += qs * ain[d];            // Q extras are zero -> only first 64 dims
  }
  const float n2 = norm2[idx];
  prow[64] = (__bf16)(0.5f - n2);        // 0.5 - norm^2
  prow[65] = (__bf16)(0.5f - n2 * n2);   // 0.5 - norm^4
#pragma unroll
  for (int d = 66; d < 96; ++d) prow[d] = (__bf16)0.0f;
#pragma unroll 6
  for (int d = 0; d < 66; ++d) arow[d] = (__bf16)ain[d];
#pragma unroll
  for (int d = 66; d < 96; ++d) arow[d] = (__bf16)0.0f;
  qdot[idx] = qd;
}

// ---------------------------------------------------------------------------
// K3: fused scores + IALSH mask + online softmax + P@V.
// One wave handles a 16-query tile of one (b,h); iterates 32 keys at a time.
// C-layout: value (r, lane) -> row = r + 8*(lane/16), col = lane%16.
// Probabilities round-trip through a wave-private LDS tile to re-shape
// C-layout -> A-layout for the P@V WMMA. V fragments are loaded early so
// their latency hides under the softmax VALU block.
// ---------------------------------------------------------------------------
__global__ void __launch_bounds__(256) attn_kernel(
    const __bf16* __restrict__ qk_bf,   // [BH, L, 64]
    const __bf16* __restrict__ qkP,     // [BH, L, 96]
    const __bf16* __restrict__ a_bf,    // [BH, L, 96]
    const __bf16* __restrict__ vT,      // [BH, 64, L]
    const float*  __restrict__ qdot,    // [BH, L]
    float* __restrict__ out) {          // [B, L, D]
  __shared__ __align__(32) __bf16 smem[8][16 * 32];

  const int lane = threadIdx.x & 31;
  const int wid  = threadIdx.x >> 5;
  const int wave = blockIdx.x * 8 + wid;
  const int st = wave & 127;           // query tile (L/16 = 128)
  const int bh = wave >> 7;            // 0..23
  const int b = bh / HH, h = bh % HH;
  const int hi = lane >> 4, cl = lane & 15;
  const int s0 = st * 16;

  const __bf16* qk = qk_bf + (size_t)bh * LL * 64;
  const __bf16* Pm = qkP   + (size_t)bh * LL * 96;
  const __bf16* am = a_bf  + (size_t)bh * LL * 96;
  const __bf16* vt = vT    + (size_t)bh * 64 * LL;
  const float*  qd = qdot  + (size_t)bh * LL;

  // Persistent A fragments: Q rows (K=64) and a rows (K=96).
  const __bf16* qrow = qk + (size_t)(s0 + cl) * 64;
  const __bf16* arow = am + (size_t)(s0 + cl) * 96;
  v16bf aq0 = load_a_frag_bf16(qrow, 0, hi);
  v16bf aq1 = load_a_frag_bf16(qrow, 32, hi);
  v16bf aa0 = load_a_frag_bf16(arow, 0, hi);
  v16bf aa1 = load_a_frag_bf16(arow, 32, hi);
  v16bf aa2 = load_a_frag_bf16(arow, 64, hi);

  v8f o0 = {}, o1 = {}, o2 = {}, o3 = {};
  float m[8], lsum[8];
#pragma unroll
  for (int r = 0; r < 8; ++r) { m[r] = -3.0e38f; lsum[r] = 0.0f; }

  __bf16* lds = &smem[wid][0];

  for (int t0 = 0; t0 < LL; t0 += 32) {
    // ---- batched fragment loads (independent regs -> clause/overlap) ----
    v16bf bk[4], bp[6], bvv[4];
    bk[0] = load_b_frag_bf16(qk, 64, t0,      0,  lane);
    bk[1] = load_b_frag_bf16(qk, 64, t0,      32, lane);
    bk[2] = load_b_frag_bf16(qk, 64, t0 + 16, 0,  lane);
    bk[3] = load_b_frag_bf16(qk, 64, t0 + 16, 32, lane);
    bp[0] = load_b_frag_bf16(Pm, 96, t0,      0,  lane);
    bp[1] = load_b_frag_bf16(Pm, 96, t0,      32, lane);
    bp[2] = load_b_frag_bf16(Pm, 96, t0,      64, lane);
    bp[3] = load_b_frag_bf16(Pm, 96, t0 + 16, 0,  lane);
    bp[4] = load_b_frag_bf16(Pm, 96, t0 + 16, 32, lane);
    bp[5] = load_b_frag_bf16(Pm, 96, t0 + 16, 64, lane);
    // V fragments now; latency hidden behind scores + softmax below.
    bvv[0] = load_b_frag_bf16(vt, LL, 0,  t0, lane);
    bvv[1] = load_b_frag_bf16(vt, LL, 16, t0, lane);
    bvv[2] = load_b_frag_bf16(vt, LL, 32, t0, lane);
    bvv[3] = load_b_frag_bf16(vt, LL, 48, t0, lane);
    const float qd0 = qd[t0 + cl];
    const float qd1 = qd[t0 + 16 + cl];

    // scores S[s,t] = qk[s].qk[t]
    v8f s0a = {}, s1a = {};
    s0a = wmma_bf16(aq0, bk[0], s0a);
    s0a = wmma_bf16(aq1, bk[1], s0a);
    s1a = wmma_bf16(aq0, bk[2], s1a);
    s1a = wmma_bf16(aq1, bk[3], s1a);
    // Pdot[s,t] = a[s] . P[t]  (K padded to 96)
    v8f p0 = {}, p1 = {};
    p0 = wmma_bf16(aa0, bp[0], p0);
    p0 = wmma_bf16(aa1, bp[1], p0);
    p0 = wmma_bf16(aa2, bp[2], p0);
    p1 = wmma_bf16(aa0, bp[3], p1);
    p1 = wmma_bf16(aa1, bp[4], p1);
    p1 = wmma_bf16(aa2, bp[5], p1);

    if (t0 + 32 < LL) {  // hint next tiles toward the caches
      __builtin_prefetch(qk + (size_t)(t0 + 32 + cl) * 64, 0, 0);
      __builtin_prefetch(Pm + (size_t)(t0 + 32 + cl) * 96, 0, 0);
      __builtin_prefetch(vt + (size_t)(cl * 4) * LL + t0 + 32, 0, 0);
    }

#pragma unroll
    for (int r = 0; r < 8; ++r) {
      float sv0 = s0a[r] * 0.125f + ((p0[r] * qd0 > 0.0f) ? 0.0f : -10000.0f);
      float sv1 = s1a[r] * 0.125f + ((p1[r] * qd1 > 0.0f) ? 0.0f : -10000.0f);
      // row max across the 16 lanes of this half-wave
      float mr = fmaxf(sv0, sv1);
#pragma unroll
      for (int mm = 1; mm < 16; mm <<= 1) mr = fmaxf(mr, __shfl_xor(mr, mm, 32));
      float mn = fmaxf(m[r], mr);
      float corr = __expf(m[r] - mn);
      m[r] = mn;
      float e0 = __expf(sv0 - mn);
      float e1 = __expf(sv1 - mn);
      float rs = e0 + e1;
#pragma unroll
      for (int mm = 1; mm < 16; mm <<= 1) rs += __shfl_xor(rs, mm, 32);
      lsum[r] = lsum[r] * corr + rs;
      o0[r] *= corr; o1[r] *= corr; o2[r] *= corr; o3[r] *= corr;
      // C-layout -> LDS tile [16 rows][32 keys] as bf16
      lds[(r + 8 * hi) * 32 + cl]      = (__bf16)e0;
      lds[(r + 8 * hi) * 32 + 16 + cl] = (__bf16)e1;
    }
    asm volatile("" ::: "memory");   // keep LDS store->load order (ds is in-order per wave)
    v16bf pf = load_a_frag_bf16(lds + cl * 32, 0, hi);
    asm volatile("" ::: "memory");
    // O += P(16x32) @ V(32x64), V fragments already resident
    o0 = wmma_bf16(pf, bvv[0], o0);
    o1 = wmma_bf16(pf, bvv[1], o1);
    o2 = wmma_bf16(pf, bvv[2], o2);
    o3 = wmma_bf16(pf, bvv[3], o3);
  }

#pragma unroll
  for (int r = 0; r < 8; ++r) {
    const int s = s0 + r + 8 * hi;
    const float inv = 1.0f / lsum[r];
    const size_t base = ((size_t)b * LL + s) * DD + h * 64;
    out[base + 0  + cl] = o0[r] * inv;
    out[base + 16 + cl] = o1[r] * inv;
    out[base + 32 + cl] = o2[r] * inv;
    out[base + 48 + cl] = o3[r] * inv;
  }
}

// ---------------------------------------------------------------------------
extern "C" void kernel_launch(void* const* d_in, const int* in_sizes, int n_in,
                              void* d_out, int out_size, void* d_ws, size_t ws_size,
                              hipStream_t stream) {
  (void)in_sizes; (void)n_in; (void)out_size; (void)ws_size;
  const float* X   = (const float*)d_in[0];  // hidden_states [B,L,D]
  const float* Wqk = (const float*)d_in[1];
  const float* bqk = (const float*)d_in[2];
  const float* Wv  = (const float*)d_in[3];
  const float* bv  = (const float*)d_in[4];
  const float* a   = (const float*)d_in[5];  // [B,H,L,66]
  float* out = (float*)d_out;

  char* ws = (char*)d_ws;
  size_t off = 0;
  auto alloc = [&](size_t bytes) -> void* {
    void* p = ws + off;
    off += (bytes + 255) & ~(size_t)255;
    return p;
  };
  __bf16* qk_bf = (__bf16*)alloc((size_t)NROW * 64 * sizeof(__bf16));  // 6.3 MB
  __bf16* vT_bf = (__bf16*)alloc((size_t)NROW * 64 * sizeof(__bf16));  // 6.3 MB
  __bf16* qkP   = (__bf16*)alloc((size_t)NROW * 96 * sizeof(__bf16));  // 9.4 MB
  __bf16* a_bf  = (__bf16*)alloc((size_t)NROW * 96 * sizeof(__bf16));  // 9.4 MB
  float*  norm2 = (float*)alloc((size_t)NROW * sizeof(float));
  float*  qdot  = (float*)alloc((size_t)NROW * sizeof(float));
  unsigned int* mmax = (unsigned int*)alloc(256);
  __bf16* Xb   = (__bf16*)alloc((size_t)BB * LL * DD * sizeof(__bf16));  // 6.3 MB
  __bf16* Wqkb = (__bf16*)alloc((size_t)DD * DD * sizeof(__bf16));       // 1.2 MB
  __bf16* Wvb  = (__bf16*)alloc((size_t)DD * DD * sizeof(__bf16));       // 1.2 MB

  init_kernel<<<1, 32, 0, stream>>>(mmax);
  {
    const int nX = BB * LL * DD, nW = DD * DD;
    cvt_bf16_kernel<<<(nX / 8 + 255) / 256, 256, 0, stream>>>(X, Xb, nX);
    cvt_bf16_kernel<<<(nW / 8 + 255) / 256, 256, 0, stream>>>(Wqk, Wqkb, nW);
    cvt_bf16_kernel<<<(nW / 8 + 255) / 256, 256, 0, stream>>>(Wv, Wvb, nW);
  }
  // 256 row-tiles * 12 heads = 3072 waves = 384 blocks of 8 waves
  proj_kernel<<<384, 256, 0, stream>>>(Xb, Wqkb, bqk, Wvb, bv,
                                       qk_bf, vT_bf, norm2, mmax);
  prep_kernel<<<(NROW + 255) / 256, 256, 0, stream>>>(a, qk_bf, norm2, mmax,
                                                      qkP, a_bf, qdot);
  // 24 (b,h) * 128 query tiles = 3072 waves = 384 blocks
  attn_kernel<<<384, 256, 0, stream>>>(qk_bf, qkP, a_bf, vT_bf, qdot, out);
}
